// AttentionHead_29910152249566
// MI455X (gfx1250) — matrix operand order repaired and checked by reading, not compiled, not run on hardware
//
#include <hip/hip_runtime.h>

typedef __attribute__((ext_vector_type(16))) _Float16 v16h;
typedef __attribute__((ext_vector_type(8)))  _Float16 v8h;
typedef __attribute__((ext_vector_type(8)))  float    v8f;

#define BB 4
#define LL 4096
#define DD 1024
#define HH 64

static __device__ __forceinline__ v8f wmma16(v16h a, v16h b, v8f c) {
  // D = A(16x32 f16) * B(32x16 f16) + C(16x16 f32)
  return __builtin_amdgcn_wmma_f32_16x16x32_f16(false, a, false, b, (short)0, c,
                                                false, false);
}

// reductions across the 16-lane half-wave (matches C-fragment row striping):
// ds_swizzle group-of-32 mode, offset = (xor_mask<<10) | and_mask(0x1f)
static __device__ __forceinline__ float red_max16(float v) {
  v = fmaxf(v, __int_as_float(__builtin_amdgcn_ds_swizzle(__float_as_int(v), 0x041f)));
  v = fmaxf(v, __int_as_float(__builtin_amdgcn_ds_swizzle(__float_as_int(v), 0x081f)));
  v = fmaxf(v, __int_as_float(__builtin_amdgcn_ds_swizzle(__float_as_int(v), 0x101f)));
  v = fmaxf(v, __int_as_float(__builtin_amdgcn_ds_swizzle(__float_as_int(v), 0x201f)));
  return v;
}
static __device__ __forceinline__ float red_sum16(float v) {
  v += __int_as_float(__builtin_amdgcn_ds_swizzle(__float_as_int(v), 0x041f));
  v += __int_as_float(__builtin_amdgcn_ds_swizzle(__float_as_int(v), 0x081f));
  v += __int_as_float(__builtin_amdgcn_ds_swizzle(__float_as_int(v), 0x101f));
  v += __int_as_float(__builtin_amdgcn_ds_swizzle(__float_as_int(v), 0x201f));
  return v;
}

// A-fragment (16x32, 16-bit): lanes 0-15 row=lane, K 0..15;
// lanes 16-31 row=lane-16, K 16..31.
static __device__ __forceinline__ v16h loadA16(const _Float16* base, int ld,
                                               int rowbase, int kbase) {
  const int lane = threadIdx.x & 31;
  const int hw = lane >> 4;
  const _Float16* p = base + (size_t)(rowbase + (lane & 15)) * ld + kbase + hw * 16;
  v8h lo = *(const v8h*)p;
  v8h hi = *(const v8h*)(p + 8);
  v16h a;
#pragma unroll
  for (int i = 0; i < 8; ++i) { a[i] = lo[i]; a[8 + i] = hi[i]; }
  return a;
}

// B-fragment (32x16, 16-bit): lanes 0-15 col=lane, K 0..15; lanes 16-31 col=lane-16,
// K 16..31. Element (k,n) = base[(colbase+n)*ld + kbase + k] -> contiguous 32B/lane.
static __device__ __forceinline__ v16h loadB16(const _Float16* base, int ld,
                                               int colbase, int kbase) {
  const int lane = threadIdx.x & 31;
  const int hw = lane >> 4;
  const _Float16* p = base + (size_t)(colbase + (lane & 15)) * ld + kbase + hw * 16;
  return *(const v16h*)p;
}

static __device__ __forceinline__ v16h loadB16c(const _Float16* base, int ld,
                                                int colbase, int colmax, int kbase) {
  const int lane = threadIdx.x & 31;
  const int hw = lane >> 4;
  int col = colbase + (lane & 15);
  col = col > colmax ? colmax : col;
  const _Float16* p = base + (size_t)col * ld + kbase + hw * 16;
  return *(const v16h*)p;
}

// ---------------- weight transpose+convert: W[D,H] f32 -> Wt[H,D] f16 ------------
__global__ void wtrans_kernel(const float* __restrict__ WQ, const float* __restrict__ WK,
                              const float* __restrict__ WV, _Float16* __restrict__ Wt) {
  int i = blockIdx.x * blockDim.x + threadIdx.x;
  if (i >= 3 * HH * DD) return;
  const int w = i / (HH * DD);
  const int rem = i - w * (HH * DD);
  const int h = rem / DD;
  const int d = rem - h * DD;
  const float* W = (w == 0) ? WQ : ((w == 1) ? WK : WV);
  Wt[i] = (_Float16)W[d * HH + h];
}

// -------- QKV projection: one wave -> 32 rows x 64 cols of ONE of {Q,K,V} --------
// Unroll-by-2 ping-pong: loads for the other buffer issue while WMMAs consume this
// one; no register-copy tail.
__global__ __launch_bounds__(256) void qkv_kernel(const float* __restrict__ idx,
                                                  const _Float16* __restrict__ Wt,
                                                  _Float16* __restrict__ Qh,
                                                  _Float16* __restrict__ Kh,
                                                  _Float16* __restrict__ Vth) {
  const int gw = blockIdx.x * 8 + (threadIdx.x >> 5);
  const int w = gw >> 9;                 // 512 row-groups per weight matrix
  const int rowbase = (gw & 511) * 32;
  const int lane = threadIdx.x & 31;
  const int hw = lane >> 4;
  const int n = lane & 15;
  const _Float16* Wb = Wt + (size_t)w * (HH * DD);

  v8f acc[2][4] = {};
  v16h Aa[2], Ba[4], Ab[2], Bb[4];

  auto loadStep = [&](int d, v16h A[2], v16h B[4]) {
#pragma unroll
    for (int rt = 0; rt < 2; ++rt) {
      const float* p = idx + (size_t)(rowbase + rt * 16 + n) * DD + d + hw * 16;
      v8f lo = *(const v8f*)p;
      v8f hi = *(const v8f*)(p + 8);
      v16h a;
#pragma unroll
      for (int i = 0; i < 8; ++i) { a[i] = (_Float16)lo[i]; a[8 + i] = (_Float16)hi[i]; }
      A[rt] = a;
    }
#pragma unroll
    for (int t = 0; t < 4; ++t) B[t] = loadB16(Wb, DD, t * 16, d);
  };
  auto compute = [&](const v16h A[2], const v16h B[4]) {
#pragma unroll
    for (int rt = 0; rt < 2; ++rt)
#pragma unroll
      for (int t = 0; t < 4; ++t)
        acc[rt][t] = wmma16(A[rt], B[t], acc[rt][t]);
  };

  loadStep(0, Aa, Ba);
  for (int d = 0; d < DD; d += 64) {
    loadStep(d + 32, Ab, Bb);            // overlaps WMMAs on buffer A
    compute(Aa, Ba);
    if (d + 64 < DD) loadStep(d + 64, Aa, Ba);  // overlaps WMMAs on buffer B
    compute(Ab, Bb);
  }

  const int b = rowbase / LL;
  const int l0 = rowbase % LL;                   // 32-row group never crosses batch
  if (w < 2) {
    _Float16* O = (w == 0) ? Qh : Kh;
    const float scale = (w == 0) ? 0.125f : 1.0f;  // fold softmax scale into Q
#pragma unroll
    for (int rt = 0; rt < 2; ++rt)
#pragma unroll
      for (int r = 0; r < 8; ++r)
#pragma unroll
        for (int t = 0; t < 4; ++t)
          O[(size_t)(rowbase + rt * 16 + r + hw * 8) * HH + t * 16 + n] =
              (_Float16)(acc[rt][t][r] * scale);
  } else {
#pragma unroll
    for (int rt = 0; rt < 2; ++rt)
#pragma unroll
      for (int r = 0; r < 8; ++r)
#pragma unroll
        for (int t = 0; t < 4; ++t)
          Vth[((size_t)b * HH + t * 16 + n) * LL + (l0 + rt * 16 + r + hw * 8)] =
              (_Float16)acc[rt][t][r];           // V stored transposed [B,H,L]
  }
}

// ---------------- causal flash attention: one wave -> 16 queries ------------------
__global__ __launch_bounds__(256) void flash_kernel(const _Float16* __restrict__ Qh,
                                                    const _Float16* __restrict__ Kh,
                                                    const _Float16* __restrict__ Vth,
                                                    float* __restrict__ out) {
  __shared__ __align__(32) _Float16 Plds[8][16 * 32];  // 1KB per wave, wave-private
  const int wslot = threadIdx.x >> 5;
  const int gw = blockIdx.x * 8 + wslot;
  const int b = gw >> 8;                 // L/16 = 256 query tiles per batch
  const int qbase = (gw & 255) * 16;
  const int lane = threadIdx.x & 31;
  const int hw = lane >> 4;
  const int n = lane & 15;

  const _Float16* Qb = Qh + (size_t)b * LL * HH;
  const _Float16* Kb = Kh + (size_t)b * LL * HH;
  const _Float16* Vb = Vth + (size_t)b * HH * LL;

  v16h A0 = loadA16(Qb, HH, qbase, 0);   // Q tile, h 0..31
  v16h A1 = loadA16(Qb, HH, qbase, 32);  // Q tile, h 32..63

  auto loadKV = [&](int kb, v16h Kf[4], v16h Vf[4]) {
    Kf[0] = loadB16c(Kb, HH, kb,      LL - 1, 0);
    Kf[1] = loadB16c(Kb, HH, kb,      LL - 1, 32);
    Kf[2] = loadB16c(Kb, HH, kb + 16, LL - 1, 0);
    Kf[3] = loadB16c(Kb, HH, kb + 16, LL - 1, 32);
#pragma unroll
    for (int t = 0; t < 4; ++t) Vf[t] = loadB16(Vb, LL, t * 16, kb);
  };

  v8f O[4] = {};
  float mrow[8], lrow[8];
#pragma unroll
  for (int r = 0; r < 8; ++r) { mrow[r] = -1e30f; lrow[r] = 0.f; }

  // process one 32-key tile; `masked` is a literal at every call site so the
  // causal-mask selects are constant-folded away for the full (pre-diagonal) tiles
  auto process = [&](int kb, const v16h Kf[4], const v16h Vf[4], bool masked) {
    // L2 prefetch 4 tiles ahead: 32 lanes cover all 32 K rows / 64 V^T rows
    const int kpf = kb + 128;
    if (kpf < qbase + 16) {
      __builtin_prefetch(Kb + (size_t)(kpf + hw * 16 + n) * HH, 0, 0);
      __builtin_prefetch(Vb + (size_t)(hw * 16 + n) * LL + kpf, 0, 0);
      __builtin_prefetch(Vb + (size_t)(32 + hw * 16 + n) * LL + kpf, 0, 0);
    }

    v8f S0 = {}, S1 = {};
    S0 = wmma16(A0, Kf[0], S0);
    S0 = wmma16(A1, Kf[1], S0);
    S1 = wmma16(A0, Kf[2], S1);
    S1 = wmma16(A1, Kf[3], S1);

#pragma unroll
    for (int r = 0; r < 8; ++r) {
      if (masked) {
        const int qi = qbase + r + hw * 8;
        S0[r] = (kb + n      <= qi) ? S0[r] : -1e30f;  // causal mask (diag tile only)
        S1[r] = (kb + 16 + n <= qi) ? S1[r] : -1e30f;
      }
      float tm = red_max16(fmaxf(S0[r], S1[r]));
      float mnew = fmaxf(mrow[r], tm);
      float fac = __expf(mrow[r] - mnew);
      mrow[r] = mnew;
      float p0 = __expf(S0[r] - mnew);
      float p1 = __expf(S1[r] - mnew);
      S0[r] = p0; S1[r] = p1;
      lrow[r] = lrow[r] * fac + red_sum16(p0 + p1);
#pragma unroll
      for (int t = 0; t < 4; ++t) O[t][r] *= fac;
    }

    // P (C-fragment) -> LDS f16 -> A-fragment for P*V
    _Float16* pl = &Plds[wslot][0];
#pragma unroll
    for (int r = 0; r < 8; ++r) {
      const int row = r + hw * 8;
      pl[row * 32 + n]      = (_Float16)S0[r];
      pl[row * 32 + 16 + n] = (_Float16)S1[r];
    }
    asm volatile("s_wait_dscnt 0x0" ::: "memory");   // wave-local LDS turnaround
    v16h Ap = loadA16(pl, 32, 0, 0);
#pragma unroll
    for (int t = 0; t < 4; ++t) O[t] = wmma16(Ap, Vf[t], O[t]);
  };

  // Exactly one tile straddles the diagonal: index nfull = qbase/32.
  // Tiles 0..nfull-1 are fully unmasked; tile nfull needs the causal mask.
  const int nfull = qbase >> 5;
  v16h K0[4], V0[4], K1[4], V1[4];
  loadKV(0, K0, V0);

  int kt = 0;
  for (; kt + 1 < nfull; kt += 2) {      // copy-free ping-pong over full tiles
    loadKV((kt + 1) * 32, K1, V1);
    process(kt * 32, K0, V0, false);
    loadKV((kt + 2) * 32, K0, V0);       // kt+2 <= nfull: tile exists (diag at nfull)
    process((kt + 1) * 32, K1, V1, false);
  }
  if (kt < nfull) {                      // odd nfull: one full tile left, then diag
    loadKV((kt + 1) * 32, K1, V1);       // kt+1 == nfull -> diagonal tile
    process(kt * 32, K0, V0, false);
    process(nfull * 32, K1, V1, true);
  } else {                               // even nfull: K0 already holds diag tile
    process(nfull * 32, K0, V0, true);
  }

#pragma unroll
  for (int r = 0; r < 8; ++r) {
    const float inv = 1.0f / lrow[r];
    const int row = qbase + r + hw * 8;
#pragma unroll
    for (int t = 0; t < 4; ++t)
      out[((size_t)b * LL + row) * HH + t * 16 + n] = O[t][r] * inv;
  }
}

extern "C" void kernel_launch(void* const* d_in, const int* in_sizes, int n_in,
                              void* d_out, int out_size, void* d_ws, size_t ws_size,
                              hipStream_t stream) {
  (void)in_sizes; (void)n_in; (void)out_size; (void)ws_size;
  const float* idx = (const float*)d_in[0];
  const float* WQ  = (const float*)d_in[1];
  const float* WK  = (const float*)d_in[2];
  const float* WV  = (const float*)d_in[3];
  float* out = (float*)d_out;

  char* ws = (char*)d_ws;
  const size_t seg = (size_t)BB * LL * HH * sizeof(_Float16);  // 2 MB
  _Float16* Qh  = (_Float16*)(ws);
  _Float16* Kh  = (_Float16*)(ws + seg);
  _Float16* Vth = (_Float16*)(ws + 2 * seg);
  _Float16* Wt  = (_Float16*)(ws + 3 * seg);   // 3 * 64*1024 f16 = 384 KB

  wtrans_kernel<<<(3 * HH * DD + 255) / 256, 256, 0, stream>>>(WQ, WK, WV, Wt);
  qkv_kernel<<<(3 * (BB * LL / 32)) / 8, 256, 0, stream>>>(idx, Wt, Qh, Kh, Vth);
  flash_kernel<<<(BB * (LL / 16)) / 8, 256, 0, stream>>>(Qh, Kh, Vth, out);
}
